// AttentionLayer_326417514821
// MI455X (gfx1250) — matrix operand order repaired and checked by reading, not compiled, not run on hardware
//
#include <hip/hip_runtime.h>

#define BATCH   8
#define S_LEN   2048
#define DMODEL  512
#define KC      32
#define NWAVE   8

typedef __attribute__((ext_vector_type(16))) __bf16 bf16x16;
typedef __attribute__((ext_vector_type(8)))  float  f32x8;
typedef __attribute__((ext_vector_type(4)))  unsigned u32x4;
typedef __attribute__((ext_vector_type(8)))  int      i32x8;
typedef __attribute__((ext_vector_type(4)))  int      i32x4;

union ABFrag {
  bf16x16  v;
  uint4    q[2];
  unsigned u[8];
};

__device__ __forceinline__ unsigned short f2bf(float f) {
  unsigned x = __float_as_uint(f);
  unsigned r = (x + 0x7fffu + ((x >> 16) & 1u)) >> 16;   // round-to-nearest-even
  return (unsigned short)r;
}
__device__ __forceinline__ unsigned pack2(float a, float b) {
  return (unsigned)f2bf(a) | ((unsigned)f2bf(b) << 16);
}

__device__ __forceinline__ f32x8 wmma_bf16(const ABFrag& a, const ABFrag& b, f32x8 c) {
  // (neg_a, A, neg_b, B, c_mod, C, reuse_a, reuse_b)
  return __builtin_amdgcn_wmma_f32_16x16x32_bf16(false, a.v, false, b.v, (short)0, c, false, false);
}

// A fragment (16x32 bf16): lane holds row (lane&15); K pair indices:
// v0..3 -> kBaseU + half*4 + {0..3},  v4..7 -> kBaseU + 8 + half*4 + {0..3}
__device__ __forceinline__ void loadA(ABFrag& f, const unsigned* rowbase, int kBaseU, int half) {
  f.q[0] = *(const uint4*)(rowbase + kBaseU + half * 4);
  f.q[1] = *(const uint4*)(rowbase + kBaseU + 8 + half * 4);
}
// B fragment (32x16 bf16): lane holds output column (lane&15); K pairs
// v0..3 -> kBaseU + half*4 + {0..3},  v4..7 -> kBaseU + 8 + half*4 + {0..3}
__device__ __forceinline__ void loadB(ABFrag& f, const unsigned* colbase, int kBaseU, int half) {
  f.q[0] = *(const uint4*)(colbase + kBaseU + half * 4);
  f.q[1] = *(const uint4*)(colbase + kBaseU + 8 + half * 4);
}

// Issue a 2-D Tensor-Data-Mover load: tile (tileDim1 rows x tileDim0 elems) of
// bf16 from a row-major tensor (row stride rowStride elems) into LDS at
// ldsAddr, optionally inserting pad dwords every padded interval (D# group1
// pad fields) so the LDS pitch avoids bank conflicts. Fields per ISA 8.3/8.4.
// This toolchain exposes the 6-argument builtin (g0, g1, g2, g3, g4, cpol).
__device__ __forceinline__ void tdm_load_2d(unsigned ldsAddr, const void* gaddr,
                                            unsigned tensorDim0, unsigned tensorDim1,
                                            unsigned tileDim0, unsigned tileDim1,
                                            unsigned rowStride,
                                            unsigned padIntervalCode,
                                            unsigned padAmountCode, bool padEnable) {
  const unsigned long long ga = (unsigned long long)gaddr;
  u32x4 g0;
  g0.x = 1u;                                               // count=1, user desc
  g0.y = ldsAddr;                                          // LDS byte address
  g0.z = (unsigned)ga;                                     // global_addr[31:0]
  g0.w = (unsigned)((ga >> 32) & 0x1FFFFFFu) | (2u << 30); // addr[56:32] | type=2
  i32x8 g1;
  g1.s0 = (int)((1u << 16) |                               // data_size = 2 bytes
                (padEnable ? (1u << 20) : 0u) |
                (padIntervalCode << 22) | (padAmountCode << 25));
  g1.s1 = (int)(tensorDim0 << 16);                         // tensor_dim0[15:0]
  g1.s2 = (int)((tensorDim0 >> 16) | (tensorDim1 << 16));  // dim0 hi | dim1 lo
  g1.s3 = (int)((tensorDim1 >> 16) | (tileDim0 << 16));    // dim1 hi | tile_dim0
  g1.s4 = (int)(tileDim1 & 0xFFFFu);                       // tile_dim1, tile_dim2=0
  g1.s5 = (int)rowStride;                                  // tensor_dim0_stride lo
  g1.s6 = 0;
  g1.s7 = 0;
  const i32x4 z4 = {0, 0, 0, 0};
  const i32x8 z8 = {0, 0, 0, 0, 0, 0, 0, 0};
  __builtin_amdgcn_tensor_load_to_lds(g0, g1, z4, z4, z8, 0);
}

// ---------------------------------------------------------------- prep kernels
__global__ __launch_bounds__(256) void pack_x_kernel(const float* __restrict__ x,
                                                     unsigned* __restrict__ out, int npairs) {
  int i = blockIdx.x * 256 + threadIdx.x;
  if (i < npairs) out[i] = pack2(x[2 * i], x[2 * i + 1]);
}

// W[DMODEL][DMODEL] fp32 (in,out) -> WT bf16 pairs laid out [out_col][in_pairs]
__global__ __launch_bounds__(256) void pack_wT_kernel(const float* __restrict__ w,
                                                      unsigned* __restrict__ out) {
  int i = blockIdx.x * 256 + threadIdx.x;  // over DMODEL*DMODEL/2
  int o = i >> 8;                          // output column
  int p = i & 255;                         // input-dim pair
  out[i] = pack2(w[(2 * p) * DMODEL + o], w[(2 * p + 1) * DMODEL + o]);
}

// ---------------------------------------------------------------- QKV GEMM
// C[M,512] = bf16(X[M,512] @ W + b). TDM double-buffered tile pipeline:
// 64x64 bf16 tiles of X and W^T staged into LDS by the Tensor Data Mover
// (pitch 36 dwords = 32 data + 4 TDM pad -> conflict-free, 16B aligned),
// 8 waves (4Mx2N), wave tile 16x32, 2 k-steps of WMMA per chunk.
__global__ __launch_bounds__(256) void qkv_gemm_kernel(
    const unsigned short* __restrict__ Xh, const unsigned short* __restrict__ WTh,
    const float* __restrict__ bias, unsigned short* __restrict__ outp) {
  __shared__ unsigned XsL[2][64][36];
  __shared__ unsigned WsL[2][64][36];

  const int tid  = threadIdx.x;
  const int wave = tid >> 5, lane = tid & 31;
  const int half = lane >> 4, r16 = lane & 15;
  const int mBlk = blockIdx.x * 64, nBlk = blockIdx.y * 64;
  const int mLoc = (wave & 3) * 16, nLoc = (wave >> 2) * 32;

  f32x8 cc[2];
  {
    const f32x8 zf = {0, 0, 0, 0, 0, 0, 0, 0};
    cc[0] = zf; cc[1] = zf;
  }

  // One TDM descriptor pair per 64-wide k-chunk. pad_interval code 4 = every
  // 32 dwords (one 64-elem bf16 row), pad_amount code 3 = 4 dwords -> 36 pitch.
  auto issue_chunk = [&](int c) {
    const int buf = c & 1;
    tdm_load_2d((unsigned)(uintptr_t)&XsL[buf][0][0],
                Xh + (size_t)mBlk * DMODEL + c * 64,
                DMODEL, 64, 64, 64, DMODEL, 4, 3, true);
    tdm_load_2d((unsigned)(uintptr_t)&WsL[buf][0][0],
                WTh + (size_t)nBlk * DMODEL + c * 64,
                DMODEL, 64, 64, 64, DMODEL, 4, 3, true);
  };

  if (wave == 0) issue_chunk(0);

  for (int c = 0; c < DMODEL / 64; ++c) {
    if (wave == 0) __builtin_amdgcn_s_wait_tensorcnt(0);
    __syncthreads();                       // chunk c resident; chunk c-1 reads done
    if (wave == 0 && c + 1 < DMODEL / 64) issue_chunk(c + 1);

    const unsigned* arow = &XsL[c & 1][mLoc + r16][0];
    const unsigned* b0r  = &WsL[c & 1][nLoc + r16][0];
    const unsigned* b1r  = &WsL[c & 1][nLoc + 16 + r16][0];
#pragma unroll
    for (int s = 0; s < 2; ++s) {
      ABFrag a, b0, b1;
      loadA(a, arow, 16 * s, half);
      loadB(b0, b0r, 16 * s, half);
      loadB(b1, b1r, 16 * s, half);
      cc[0] = wmma_bf16(a, b0, cc[0]);
      cc[1] = wmma_bf16(a, b1, cc[1]);
    }
  }

#pragma unroll
  for (int t = 0; t < 2; ++t) {
    const int col = nBlk + nLoc + 16 * t + r16;
    const float bv = bias[col];
#pragma unroll
    for (int r = 0; r < 8; ++r) {
      const int row = mBlk + mLoc + r + 8 * half;
      outp[(size_t)row * DMODEL + col] = f2bf(cc[t][r] + bv);
    }
  }
}

// ------------------------------------------------- flash attention + layernorm
// One block: 16 query rows x full D. 8 waves split d_key (QK^T partials) and
// d_val (64 output cols each). Online softmax by wave 0, P/V shared via LDS.
__global__ __launch_bounds__(256) void attn_ln_kernel(
    const unsigned* __restrict__ Qu, const unsigned* __restrict__ Ku,
    const unsigned short* __restrict__ Vh, const unsigned char* __restrict__ mask,
    const float* __restrict__ gamma, const float* __restrict__ beta,
    float* __restrict__ out) {
  __shared__ float    Spart[NWAVE][16][33];  // partial S tiles (padded)
  __shared__ unsigned PsL[16][17];           // P tile, bf16 pairs (padded)
  __shared__ unsigned VsL[DMODEL][17];       // V^T tile, bf16 key-pairs (padded)
  __shared__ float alphaS[16], linvS[16], rowsum[16], rowsq[16];

  const int tid  = threadIdx.x;
  const int wave = tid >> 5, lane = tid & 31;
  const int half = lane >> 4, r16 = lane & 15;
  const int b  = blockIdx.y;
  const int q0 = blockIdx.x * 16;
  const int dw = wave * 64;                      // this wave's d_val / d_key slice
  const float scale = 0.044194173824159216f;     // 1/sqrt(512)

  // Preload this wave's Q fragments (16 rows x d[dw..dw+64), 2 k-steps).
  ABFrag aQ[2];
  {
    const unsigned* rowbase = Qu + (size_t)(b * S_LEN + q0 + r16) * (DMODEL / 2);
    loadA(aQ[0], rowbase, dw / 2, half);
    loadA(aQ[1], rowbase, dw / 2 + 16, half);
  }

  f32x8 acc[4];
  {
    const f32x8 zf = {0, 0, 0, 0, 0, 0, 0, 0};
#pragma unroll
    for (int t = 0; t < 4; ++t) acc[t] = zf;
  }
  float m_i = -1e30f, l_i = 0.0f;

  for (int kb = 0; kb < S_LEN; kb += KC) {
    if (kb + KC < S_LEN) {  // pull next tiles toward L2/L0
      __builtin_prefetch(Vh + (size_t)(b * S_LEN + kb + KC) * DMODEL, 0, 1);
      __builtin_prefetch(Ku + (size_t)(b * S_LEN + kb + KC) * (DMODEL / 2), 0, 1);
    }

    // Stage V tile transposed: VsL[d][p] = pack(V[kb+2p][d], V[kb+2p+1][d])
    for (int i = tid; i < DMODEL * (KC / 2); i += 256) {
      const int d = i & (DMODEL - 1);
      const int p = i >> 9;  // 0..15
      const size_t base = (size_t)(b * S_LEN + kb + 2 * p) * DMODEL + d;
      VsL[d][p] = (unsigned)Vh[base] | ((unsigned)Vh[base + DMODEL] << 16);
    }

    // Partial S = Q[:, dw:dw+64] @ K[:, dw:dw+64]^T  (two 16x16 tiles)
    f32x8 s0, s1;
    {
      const f32x8 zf = {0, 0, 0, 0, 0, 0, 0, 0};
      s0 = zf; s1 = zf;
    }
    const unsigned* kc0 = Ku + (size_t)(b * S_LEN + kb + r16)      * (DMODEL / 2);
    const unsigned* kc1 = Ku + (size_t)(b * S_LEN + kb + 16 + r16) * (DMODEL / 2);
#pragma unroll
    for (int ks = 0; ks < 2; ++ks) {
      ABFrag bK0, bK1;
      loadB(bK0, kc0, dw / 2 + 16 * ks, half);
      loadB(bK1, kc1, dw / 2 + 16 * ks, half);
      s0 = wmma_bf16(aQ[ks], bK0, s0);
      s1 = wmma_bf16(aQ[ks], bK1, s1);
    }
#pragma unroll
    for (int r = 0; r < 8; ++r) {
      Spart[wave][r + 8 * half][r16]      = s0[r];
      Spart[wave][r + 8 * half][16 + r16] = s1[r];
    }
    __syncthreads();

    // Online softmax (wave 0, one lane per query row)
    if (wave == 0 && lane < 16) {
      const int row = lane;
      float sv[KC];
#pragma unroll
      for (int j = 0; j < KC; ++j) {
        float s = 0.0f;
#pragma unroll
        for (int w2 = 0; w2 < NWAVE; ++w2) s += Spart[w2][row][j];
        s *= scale;
        const unsigned char mk = mask[((size_t)b * S_LEN + (kb + j)) * S_LEN + (q0 + row)];
        sv[j] = mk ? s : -1e9f;
      }
      float mnew = m_i;
#pragma unroll
      for (int j = 0; j < KC; ++j) mnew = fmaxf(mnew, sv[j]);
      const float alpha = __expf(m_i - mnew);
      float ln = l_i * alpha;
#pragma unroll
      for (int j = 0; j < KC; j += 2) {
        const float p0 = __expf(sv[j] - mnew);
        const float p1 = __expf(sv[j + 1] - mnew);
        ln += p0 + p1;
        PsL[row][j >> 1] = pack2(p0, p1);
      }
      m_i = mnew; l_i = ln;
      alphaS[row] = alpha;
    }
    __syncthreads();

    // Rescale running O and accumulate P @ V[:, dw:dw+64]
    float arow[8];
#pragma unroll
    for (int r = 0; r < 8; ++r) arow[r] = alphaS[r + 8 * half];

    ABFrag pA;
#pragma unroll
    for (int v = 0; v < 4; ++v) pA.u[v]     = PsL[r16][half * 4 + v];
#pragma unroll
    for (int v = 0; v < 4; ++v) pA.u[4 + v] = PsL[r16][8 + half * 4 + v];

#pragma unroll
    for (int t = 0; t < 4; ++t) {
      ABFrag bV;
      const int d = dw + 16 * t + r16;
#pragma unroll
      for (int v = 0; v < 8; ++v) bV.u[v] = VsL[d][(v >> 2) * 8 + half * 4 + (v & 3)];
#pragma unroll
      for (int r = 0; r < 8; ++r) acc[t][r] = acc[t][r] * arow[r];
      acc[t] = wmma_bf16(pA, bV, acc[t]);
    }
    __syncthreads();
  }

  if (wave == 0 && lane < 16) linvS[lane] = 1.0f / l_i;
  if (tid < 16) { rowsum[tid] = 0.0f; rowsq[tid] = 0.0f; }
  __syncthreads();

  // Normalize by l, then LayerNorm stats via shared-memory float atomics
  float lr[8], ps[8], ps2[8];
#pragma unroll
  for (int r = 0; r < 8; ++r) { lr[r] = linvS[r + 8 * half]; ps[r] = 0.0f; ps2[r] = 0.0f; }
#pragma unroll
  for (int t = 0; t < 4; ++t)
#pragma unroll
    for (int r = 0; r < 8; ++r) {
      acc[t][r] = acc[t][r] * lr[r];
      const float v = acc[t][r];
      ps[r] += v; ps2[r] += v * v;
    }
#pragma unroll
  for (int r = 0; r < 8; ++r) {
    atomicAdd(&rowsum[r + 8 * half], ps[r]);
    atomicAdd(&rowsq[r + 8 * half], ps2[r]);
  }
  __syncthreads();

  const float invD = 1.0f / (float)DMODEL;
#pragma unroll
  for (int t = 0; t < 4; ++t) {
    const int col = dw + 16 * t + r16;
    const float g  = gamma[col];
    const float be = beta[col];
#pragma unroll
    for (int r = 0; r < 8; ++r) {
      const int row = r + 8 * half;
      const float mu  = rowsum[row] * invD;
      const float var = rowsq[row] * invD - mu * mu;
      const float rs  = rsqrtf(var + 1e-5f);
      out[(size_t)(b * S_LEN + q0 + row) * DMODEL + col] = (acc[t][r] - mu) * rs * g + be;
    }
  }
}

// ------------------------------------------------------------------- launcher
extern "C" void kernel_launch(void* const* d_in, const int* in_sizes, int n_in,
                              void* d_out, int out_size, void* d_ws, size_t ws_size,
                              hipStream_t stream) {
  (void)in_sizes; (void)n_in; (void)out_size; (void)ws_size;
  const float*         X    = (const float*)d_in[0];
  const unsigned char* mask = (const unsigned char*)d_in[1];
  const float* Wq = (const float*)d_in[2];
  const float* bq = (const float*)d_in[3];
  const float* Wk = (const float*)d_in[4];
  const float* bk = (const float*)d_in[5];
  const float* Wv = (const float*)d_in[6];
  const float* bv = (const float*)d_in[7];
  const float* gamma = (const float*)d_in[8];
  const float* beta  = (const float*)d_in[9];
  float* out = (float*)d_out;

  const size_t M = (size_t)BATCH * S_LEN;  // 16384
  unsigned short* Xbf = (unsigned short*)d_ws;               // M*D bf16
  unsigned short* WT  = Xbf + M * DMODEL;                    // 3*D*D bf16 (transposed)
  unsigned short* Qb  = WT + (size_t)3 * DMODEL * DMODEL;    // M*D bf16
  unsigned short* Kb  = Qb + M * DMODEL;
  unsigned short* Vb  = Kb + M * DMODEL;

  const int xpairs = (int)(M * DMODEL / 2);
  pack_x_kernel<<<dim3((xpairs + 255) / 256), 256, 0, stream>>>(X, (unsigned*)Xbf, xpairs);
  const int wpairs = DMODEL * DMODEL / 2;
  pack_wT_kernel<<<dim3(wpairs / 256), 256, 0, stream>>>(Wq, (unsigned*)WT);
  pack_wT_kernel<<<dim3(wpairs / 256), 256, 0, stream>>>(Wk, (unsigned*)(WT + (size_t)DMODEL * DMODEL));
  pack_wT_kernel<<<dim3(wpairs / 256), 256, 0, stream>>>(Wv, (unsigned*)(WT + (size_t)2 * DMODEL * DMODEL));

  dim3 ggrid((unsigned)(M / 64), DMODEL / 64);
  qkv_gemm_kernel<<<ggrid, 256, 0, stream>>>(Xbf, WT, bq, Qb);
  qkv_gemm_kernel<<<ggrid, 256, 0, stream>>>(Xbf, WT + (size_t)DMODEL * DMODEL, bk, Kb);
  qkv_gemm_kernel<<<ggrid, 256, 0, stream>>>(Xbf, WT + (size_t)2 * DMODEL * DMODEL, bv, Vb);

  attn_ln_kernel<<<dim3(S_LEN / 16, BATCH), 256, 0, stream>>>(
      (const unsigned*)Qb, (const unsigned*)Kb, (const unsigned short*)Vb,
      mask, gamma, beta, out);
}
